// CRF_61770219651514
// MI455X (gfx1250) — compile-verified, matrix operand order checked by reading, not run on hardware
//
#include <hip/hip_runtime.h>
#include <cstdint>

#define Bsz  512
#define Tlen 1024
#define Ktag 64

typedef __attribute__((ext_vector_type(16))) _Float16 v16h;
typedef __attribute__((ext_vector_type(8)))  _Float16 v8h;
typedef __attribute__((ext_vector_type(8)))  float    v8f;

// ---------------------------------------------------------------------------
// Forward-algorithm log-partition, scaled-probability formulation.
// One wave (32 lanes) owns 16 batch rows; grid = 32 waves covers B=512.
// Per scan step: p_new[b,j] = (sum_i p[b,i]*expT[i,j]) * exp(e[b,j]),
// renormalized so max_j p = 1, with log-offset c[b] accumulating the scale.
// The inner contraction is 8x v_wmma_f32_16x16x32_f16 per step.
// ---------------------------------------------------------------------------
__global__ __launch_bounds__(32)
void crf_partition_kernel(const float* __restrict__ feat,   // [B,T,K]
                          const float* __restrict__ trans,  // [K,K]
                          const uint8_t* __restrict__ mask, // [B,T] bool
                          float* __restrict__ logZ)         // [B]
{
  const int lane = threadIdx.x & 31;
  const int half = lane >> 4;   // 0/1 -> rows r / r+8 in C layout
  const int col  = lane & 15;   // N (tag) column, and A-operand row M
  const int b0   = blockIdx.x * 16;

  __shared__ __align__(16) _Float16 Ebuf[16 * 64]; // A staging: [row][K] f16

  // ---- expT resident in VGPRs, WMMA-B layout (mirror of the 16-bit A table:
  //      v16h element e -> VGPR e/2, f16 slot e&1; K = 2*(e/2)+ (e&1) + 8*half,
  //      +16 for e>=8; N = lane%16). 2 K-chunks x 4 N-tiles = 64 VGPRs. ----
  v16h Bop[2][4];
#pragma unroll
  for (int c = 0; c < 2; ++c)
#pragma unroll
    for (int n = 0; n < 4; ++n)
#pragma unroll
      for (int e = 0; e < 16; ++e) {
        const int v  = e >> 1;
        const int kk = (v < 4) ? (2 * v + (e & 1) + 8 * half)
                               : (16 + 2 * (v - 4) + (e & 1) + 8 * half);
        const int krow = 32 * c + kk;
        Bop[c][n][e] = (_Float16)__expf(trans[krow * Ktag + 16 * n + col]);
      }

  // ---- state: p in C layout (4 tiles x 8 rows), per-row log offset c ----
  float p[4][8];
  float cacc[8];
  float ef[4][8];

  // t = 0: alpha = feat[:,0,:]  ->  p = exp(f0)/m, c = log m
#pragma unroll
  for (int n = 0; n < 4; ++n)
#pragma unroll
    for (int r = 0; r < 8; ++r) {
      const size_t row = (size_t)(b0 + r + 8 * half);
      ef[n][r] = __expf(feat[row * Tlen * Ktag + 16 * n + col]);
    }
  {
    float m[8];
#pragma unroll
    for (int r = 0; r < 8; ++r)
      m[r] = fmaxf(fmaxf(ef[0][r], ef[1][r]), fmaxf(ef[2][r], ef[3][r]));
#pragma unroll
    for (int sh = 8; sh >= 1; sh >>= 1)
#pragma unroll
      for (int r = 0; r < 8; ++r)
        m[r] = fmaxf(m[r], __shfl_xor(m[r], sh, 32));
#pragma unroll
    for (int r = 0; r < 8; ++r) {
      const float inv = __builtin_amdgcn_rcpf(m[r]);
      cacc[r] = -__logf(inv);            // == log(m) consistent with applied scale
#pragma unroll
      for (int n = 0; n < 4; ++n) p[n][r] = ef[n][r] * inv;
    }
  }

  for (int t = 1; t < Tlen; ++t) {
    // deep prefetch of emissions (distance 8 steps; dedupes to cachelines)
    if (t + 8 < Tlen) {
#pragma unroll
      for (int r = 0; r < 8; ++r) {
        const size_t row = (size_t)(b0 + r + 8 * half);
        const float* base = feat + row * Tlen * Ktag + (size_t)(t + 8) * Ktag;
        __builtin_prefetch(base + col, 0, 0);
        __builtin_prefetch(base + 32 + col, 0, 0);
      }
    }

    // emission loads + exp (off the dependent chain; overlaps LDS/WMMA below)
#pragma unroll
    for (int n = 0; n < 4; ++n)
#pragma unroll
      for (int r = 0; r < 8; ++r) {
        const size_t row = (size_t)(b0 + r + 8 * half);
        ef[n][r] = __expf(feat[row * Tlen * Ktag + (size_t)t * Ktag + 16 * n + col]);
      }
    uint8_t mk[8];
#pragma unroll
    for (int r = 0; r < 8; ++r)
      mk[r] = mask[(size_t)(b0 + r + 8 * half) * Tlen + t];

    // ---- p (C layout, f32) -> f16 A staging in LDS, row-major [row][K] ----
#pragma unroll
    for (int n = 0; n < 4; ++n)
#pragma unroll
      for (int r = 0; r < 8; ++r)
        Ebuf[(r + 8 * half) * 64 + 16 * n + col] = (_Float16)p[n][r];
    // Single wave: LDS executes in order, so RAW through Ebuf is guaranteed by
    // hardware. wave_barrier is a zero-cost compiler fence (no dscnt-0 drain).
    __builtin_amdgcn_wave_barrier();

    // A-layout reload: per chunk two contiguous 16B rows (ds_load_b128)
    v16h A[2];
#pragma unroll
    for (int c = 0; c < 2; ++c) {
      const v8h lo = *(const v8h*)&Ebuf[col * 64 + 32 * c + 8 * half];
      const v8h hi = *(const v8h*)&Ebuf[col * 64 + 32 * c + 8 * half + 16];
      A[c] = __builtin_shufflevector(lo, hi, 0, 1, 2, 3, 4, 5, 6, 7,
                                     8, 9, 10, 11, 12, 13, 14, 15);
    }
    __builtin_amdgcn_wave_barrier();  // WAR fence vs next iteration's stores

    // ---- S = p x expT : four 16x16 f32 tiles, K=64 in two x32 chunks ----
    v8f acc[4];
#pragma unroll
    for (int n = 0; n < 4; ++n) {
      v8f z = {};
      z = __builtin_amdgcn_wmma_f32_16x16x32_f16(false, A[0], false, Bop[0][n],
                                                 (short)0, z, false, false);
      z = __builtin_amdgcn_wmma_f32_16x16x32_f16(false, A[1], false, Bop[1][n],
                                                 (short)0, z, false, false);
      acc[n] = z;
    }

    // ---- s = S * exp(e); row renormalize; masked update ----
    float s[4][8];
#pragma unroll
    for (int n = 0; n < 4; ++n)
#pragma unroll
      for (int r = 0; r < 8; ++r)
        s[n][r] = acc[n][r] * ef[n][r];

    float m[8];
#pragma unroll
    for (int r = 0; r < 8; ++r)
      m[r] = fmaxf(fmaxf(s[0][r], s[1][r]), fmaxf(s[2][r], s[3][r]));
#pragma unroll
    for (int sh = 8; sh >= 1; sh >>= 1)
#pragma unroll
      for (int r = 0; r < 8; ++r)
        m[r] = fmaxf(m[r], __shfl_xor(m[r], sh, 32));

#pragma unroll
    for (int r = 0; r < 8; ++r) {
      const float inv = __builtin_amdgcn_rcpf(m[r]);
      const bool um = (mk[r] != 0);
#pragma unroll
      for (int n = 0; n < 4; ++n)
        p[n][r] = um ? s[n][r] * inv : p[n][r];
      cacc[r] = um ? cacc[r] - __logf(inv) : cacc[r];
    }
  }

  // logZ[b] = c[b] + log(sum_j p[b,j])
  float ts[8];
#pragma unroll
  for (int r = 0; r < 8; ++r)
    ts[r] = (p[0][r] + p[1][r]) + (p[2][r] + p[3][r]);
#pragma unroll
  for (int sh = 8; sh >= 1; sh >>= 1)
#pragma unroll
    for (int r = 0; r < 8; ++r)
      ts[r] += __shfl_xor(ts[r], sh, 32);
  if (col == 0) {
#pragma unroll
    for (int r = 0; r < 8; ++r)
      logZ[b0 + r + 8 * half] = cacc[r] + __logf(ts[r]);
  }
}

// ---------------------------------------------------------------------------
// Gold-path score (thread per batch) + final subtraction.
// last_tag = tag at most recent masked position <= t-1 (default position 0).
// ---------------------------------------------------------------------------
__global__ void crf_score_kernel(const float* __restrict__ feat,
                                 const float* __restrict__ trans,
                                 const int* __restrict__ tag,
                                 const uint8_t* __restrict__ mask,
                                 const float* __restrict__ logZ,
                                 float* __restrict__ out)
{
  const int b = blockIdx.x * blockDim.x + threadIdx.x;
  if (b >= Bsz) return;
  const float* fb = feat + (size_t)b * Tlen * Ktag;
  const int*   tb = tag + (size_t)b * Tlen;
  const uint8_t* mb = mask + (size_t)b * Tlen;

  int last = tb[0];
  float s = fb[last];                       // emit[:,0] (mask[:,0] forced True)
  for (int t = 1; t < Tlen; ++t) {
    const int tg = tb[t];
    if (mb[t]) {
      s += fb[(size_t)t * Ktag + tg] + trans[last * Ktag + tg];
      last = tg;                            // becomes last_tag for t' > t
    }
  }
  out[b] = s - logZ[b];
}

extern "C" void kernel_launch(void* const* d_in, const int* in_sizes, int n_in,
                              void* d_out, int out_size, void* d_ws, size_t ws_size,
                              hipStream_t stream) {
  const float*   feat  = (const float*)d_in[0];   // [512,1024,64] f32
  const float*   trans = (const float*)d_in[1];   // [64,64] f32
  const int*     tag   = (const int*)d_in[2];     // [512,1024] i32
  const uint8_t* mask  = (const uint8_t*)d_in[3]; // [512,1024] bool
  float* out  = (float*)d_out;                    // [512] f32
  float* logZ = (float*)d_ws;                     // scratch: 512 f32

  crf_partition_kernel<<<Bsz / 16, 32, 0, stream>>>(feat, trans, mask, logZ);
  crf_score_kernel<<<(Bsz + 255) / 256, 256, 0, stream>>>(feat, trans, tag, mask,
                                                          logZ, out);
}
